// Attention_38405597560936
// MI455X (gfx1250) — matrix operand order repaired and checked by reading, not compiled, not run on hardware
//
#include <hip/hip_runtime.h>
#include <hip/hip_bf16.h>

// ---------------------------------------------------------------------------
// MI455X (gfx1250, wave32) fused GroupNorm + attention block.
// GEMMs on v_wmma_f32_16x16x32_bf16 (f32 accumulate); attention K/V slabs are
// staged in LDS via global_load_async_to_lds_b128 (ASYNCcnt), double-buffered.
// ---------------------------------------------------------------------------

typedef __attribute__((ext_vector_type(16))) __bf16 v16bf;
typedef __attribute__((ext_vector_type(8)))  __bf16 v8bf;
typedef __attribute__((ext_vector_type(8)))  float  v8f;

#define WMMA_BF16(A, B, C) \
  __builtin_amdgcn_wmma_f32_16x16x32_bf16(false, (A), false, (B), (short)0, (C), false, false)

constexpr int CB = 2;            // batch
constexpr int CC = 512;          // channels
constexpr int CN = 48 * 48;      // spatial positions = 2304
constexpr int NHEADS = 8;
constexpr int DHEAD = 64;
constexpr int NGROUPS = 32;
constexpr int GCH = CC / NGROUPS;   // 16 channels per group

// A-matrix (16x32 bf16) per-lane fragment: lane m = lane&15, half = lane>>4.
// elems 0..15 -> K = half*16 + e (contiguous, matches B)
static __device__ __forceinline__ v16bf load_a_frag(const __bf16* __restrict__ row, int half) {
  v8bf lo = *(const v8bf*)(row + half * 16);
  v8bf hi = *(const v8bf*)(row + half * 16 + 8);
  return __builtin_shufflevector(lo, hi, 0,1,2,3,4,5,6,7,8,9,10,11,12,13,14,15);
}
// B-matrix (32x16 bf16): lane n = lane&15, K = (lane>>4)*16 + e -> one 32B load.

// Per-lane async copy of 16 bytes global -> LDS (tracked by ASYNCcnt).
static __device__ __forceinline__ void async_copy_b128(const void* gptr, void* lptr) {
  unsigned lds_off = (unsigned)(size_t)lptr;   // flat LDS addr low 32 bits = LDS offset
  asm volatile("global_load_async_to_lds_b128 %0, %1, off"
               :: "v"(lds_off), "v"(gptr) : "memory");
}

// ---------------------------------------------------------------------------
// Kernel 1: GroupNorm statistics. One block per (b, group); 16*2304 elements.
// ---------------------------------------------------------------------------
__global__ __launch_bounds__(256) void gn_stats(const float* __restrict__ x,
                                                float* __restrict__ stats) {
  __shared__ float ssum[256], ssq[256];
  const int bg = blockIdx.x;                       // 0..63 (groups are channel-contiguous)
  const float* xp = x + (size_t)bg * (GCH * CN);
  float s = 0.f, q = 0.f;
  for (int i = threadIdx.x; i < GCH * CN; i += 256) {
    float v = xp[i];
    s += v; q += v * v;
  }
  ssum[threadIdx.x] = s; ssq[threadIdx.x] = q;
  __syncthreads();
  for (int st = 128; st > 0; st >>= 1) {
    if ((int)threadIdx.x < st) {
      ssum[threadIdx.x] += ssum[threadIdx.x + st];
      ssq[threadIdx.x]  += ssq[threadIdx.x + st];
    }
    __syncthreads();
  }
  if (threadIdx.x == 0) {
    const float inv_n = 1.0f / (float)(GCH * CN);
    float mean = ssum[0] * inv_n;
    float var  = ssq[0] * inv_n - mean * mean;
    stats[bg * 2 + 0] = mean;
    stats[bg * 2 + 1] = rsqrtf(var + 1e-5f);
  }
}

// ---------------------------------------------------------------------------
// Kernel 2: apply GroupNorm, write Xt[b][n][c] (c contiguous) in bf16.
// ---------------------------------------------------------------------------
__global__ __launch_bounds__(256) void gn_apply(const float* __restrict__ x,
                                                const float* __restrict__ w,
                                                const float* __restrict__ bvec,
                                                const float* __restrict__ stats,
                                                __bf16* __restrict__ Xt) {
  size_t idx = (size_t)blockIdx.x * blockDim.x + threadIdx.x;
  if (idx >= (size_t)CB * CN * CC) return;
  int c   = (int)(idx % CC);
  size_t t = idx / CC;
  int pos = (int)(t % CN);
  int b   = (int)(t / CN);
  int bg  = b * NGROUPS + (c >> 4);
  float mean = stats[bg * 2 + 0];
  float rstd = stats[bg * 2 + 1];
  float v = x[((size_t)b * CC + c) * CN + pos];
  Xt[idx] = (__bf16)((v - mean) * rstd * w[c] + bvec[c]);
}

// ---------------------------------------------------------------------------
// Kernel 3: fp32 -> bf16 weight conversion (qkv_w 1536x512, proj_w 512x512).
// ---------------------------------------------------------------------------
__global__ __launch_bounds__(256) void cvt_w(const float* __restrict__ wq,
                                             const float* __restrict__ wp,
                                             __bf16* __restrict__ wqb,
                                             __bf16* __restrict__ wpb) {
  int i = blockIdx.x * blockDim.x + threadIdx.x;
  if (i < 3 * CC * CC) wqb[i] = (__bf16)wq[i];
  if (i < CC * CC)     wpb[i] = (__bf16)wp[i];
}

// ---------------------------------------------------------------------------
// Kernel 4: QKV GEMM. y[o,n] = sum_c W[o,c] * Xt[n,c], + bias, scattered into
// q,k (as [head][n][d], k pre-scaled by 1/sqrt(d)) and v (as [head][d][n]).
// ---------------------------------------------------------------------------
__global__ __launch_bounds__(128) void qkv_gemm(const __bf16* __restrict__ W,
                                                const __bf16* __restrict__ Xt,
                                                const float* __restrict__ bias,
                                                __bf16* __restrict__ qb,
                                                __bf16* __restrict__ kb,
                                                __bf16* __restrict__ vb) {
  const int lane = threadIdx.x & 31, wave = threadIdx.x >> 5;
  const int half = lane >> 4, l16 = lane & 15;
  const int nbase = blockIdx.x * 16;                  // 144 position tiles
  const int obase = (blockIdx.y * 4 + wave) * 16;     // 96 output-channel tiles
  const int b = blockIdx.z;

  const __bf16* wrow = W + (size_t)(obase + l16) * CC;                 // A: row m = l16
  const __bf16* xrow = Xt + ((size_t)b * CN + nbase + l16) * CC;       // B: col n = l16
  v8f acc = {};
#pragma unroll
  for (int c = 0; c < CC; c += 32) {
    v16bf A = load_a_frag(wrow + c, half);
    v16bf B = *(const v16bf*)(xrow + c + half * 16);
    acc = WMMA_BF16(A, B, acc);
  }
#pragma unroll
  for (int r = 0; r < 8; ++r) {
    int o   = obase + r + 8 * half;
    int pos = nbase + l16;
    float val = acc[r] + bias[o];
    int h = o / (3 * DHEAD);            // head within batch
    int rem = o % (3 * DHEAD);
    int i = rem / 3, s = rem % 3;       // channel-in-head, q/k/v selector
    size_t head = (size_t)b * NHEADS + h;
    if (s == 0)      qb[(head * CN + pos) * DHEAD + i] = (__bf16)val;
    else if (s == 1) kb[(head * CN + pos) * DHEAD + i] = (__bf16)(val * 0.125f); // 1/sqrt(64)
    else             vb[(head * DHEAD + i) * CN + pos] = (__bf16)val;
  }
}

// ---------------------------------------------------------------------------
// Kernel 5: flash attention. grid = (36 qtile-groups, 16 heads), 4 waves/block.
// Each wave owns 16 queries; the block shares one head, so each 32-key slab of
// K (32x64 bf16) and V (64x32 bf16) is DMA'd into LDS ONCE per block with
// global_load_async_to_lds_b128, double-buffered, and consumed by all 4 waves.
// ---------------------------------------------------------------------------
__global__ __launch_bounds__(128) void attn(const __bf16* __restrict__ qg,
                                            const __bf16* __restrict__ kg,
                                            const __bf16* __restrict__ vg,
                                            __bf16* __restrict__ At) {
  __shared__ __bf16 kvbuf[2][2][2048];       // [buf][0=K 32x64 | 1=V 64x32], 4KB each
  __shared__ __bf16 pstage[4][16 * 32];      // per-wave P-tile relayout buffer
  const int lane = threadIdx.x & 31, wave = threadIdx.x >> 5;
  const int half = lane >> 4, l16 = lane & 15;
  const int head = blockIdx.y;
  const int b = head >> 3, hh = head & 7;
  const int qbase = (blockIdx.x * 4 + wave) * 16;

  const __bf16* qh = qg + (size_t)head * CN * DHEAD;
  const __bf16* kh = kg + (size_t)head * CN * DHEAD;
  const __bf16* vh = vg + (size_t)head * DHEAD * CN;
  __bf16* pl = &pstage[wave][0];

  // Cooperative async copy of one 32-key slab: 4 x b128 per thread (2 K + 2 V).
  auto issue_slab = [&](int buf, int kbp) {
#pragma unroll
    for (int u = 0; u < 2; ++u) {
      int c = (int)threadIdx.x + u * 128;          // chunk id, 0..255 (16B chunks)
      const __bf16* gk = kh + (size_t)(kbp + (c >> 3)) * DHEAD + (c & 7) * 8;
      async_copy_b128(gk, &kvbuf[buf][0][c * 8]);
      const __bf16* gv = vh + (size_t)(c >> 2) * CN + kbp + (c & 3) * 8;
      async_copy_b128(gv, &kvbuf[buf][1][c * 8]);
    }
  };

  // Q fragments for K-dim c=0..31 and c=32..63 (row m = l16, contiguous d=64)
  const __bf16* qrow = qh + (size_t)(qbase + l16) * DHEAD;
  v16bf Aq0 = load_a_frag(qrow, half);
  v16bf Aq1 = load_a_frag(qrow + 32, half);

  float mrow[8], lrow[8];
  v8f acc[4] = {};
#pragma unroll
  for (int r = 0; r < 8; ++r) { mrow[r] = -3.0e38f; lrow[r] = 0.f; }

  issue_slab(0, 0);
  constexpr int NIT = CN / 32;   // 72
  for (int it = 0; it < NIT; ++it) {
    const int cur = it & 1;
    if (it + 1 < NIT) {
      issue_slab(cur ^ 1, (it + 1) * 32);
      asm volatile("s_wait_asynccnt 0x4" ::: "memory");  // current slab landed
    } else {
      asm volatile("s_wait_asynccnt 0x0" ::: "memory");
    }
    __syncthreads();                                     // slab visible to all waves
    const __bf16* kbuf = &kvbuf[cur][0][0];
    const __bf16* vbuf = &kvbuf[cur][1][0];

    // ---- scores: two 16x16 tiles over this 32-key slab (K = channel dim 64) ----
    v8f S0 = {}, S1 = {};
    S0 = WMMA_BF16(Aq0, *(const v16bf*)(kbuf + l16 * DHEAD + half * 16), S0);
    S0 = WMMA_BF16(Aq1, *(const v16bf*)(kbuf + l16 * DHEAD + 32 + half * 16), S0);
    S1 = WMMA_BF16(Aq0, *(const v16bf*)(kbuf + (16 + l16) * DHEAD + half * 16), S1);
    S1 = WMMA_BF16(Aq1, *(const v16bf*)(kbuf + (16 + l16) * DHEAD + 32 + half * 16), S1);

    // ---- online softmax across the 32 keys (rows live on 16-lane groups) ----
#pragma unroll
    for (int r = 0; r < 8; ++r) {
      float t = fmaxf(S0[r], S1[r]);
#pragma unroll
      for (int m = 8; m >= 1; m >>= 1) t = fmaxf(t, __shfl_xor(t, m, 16));
      float mn = fmaxf(mrow[r], t);
      float p0 = __expf(S0[r] - mn);
      float p1 = __expf(S1[r] - mn);
      float s = p0 + p1;
#pragma unroll
      for (int m = 8; m >= 1; m >>= 1) s += __shfl_xor(s, m, 16);
      float alpha = __expf(mrow[r] - mn);
      lrow[r] = lrow[r] * alpha + s;
      mrow[r] = mn;
      acc[0][r] *= alpha; acc[1][r] *= alpha; acc[2][r] *= alpha; acc[3][r] *= alpha;
      int mq = r + 8 * half;                       // C-layout row -> query index
      pl[mq * 32 + l16]      = (__bf16)p0;
      pl[mq * 32 + 16 + l16] = (__bf16)p1;
    }

    // ---- re-read P as a 16x32 A fragment (same wave; DS ops are in-order) ----
    v16bf Ap = load_a_frag(pl + l16 * 32, half);

    // ---- PV: 4 output-channel tiles; B = V slab rows (K = key dim, contiguous) ----
#pragma unroll
    for (int ct = 0; ct < 4; ++ct) {
      v16bf Bv = *(const v16bf*)(vbuf + (ct * 16 + l16) * 32 + half * 16);
      acc[ct] = WMMA_BF16(Ap, Bv, acc[ct]);
    }
    __syncthreads();                 // all waves done with buf[cur] before refill
  }

  // ---- normalize and write At[b][pos][channel] (c contiguous) ----
#pragma unroll
  for (int r = 0; r < 8; ++r) {
    float inv = 1.0f / lrow[r];
    int pos = qbase + r + 8 * half;
#pragma unroll
    for (int ct = 0; ct < 4; ++ct) {
      int ch = hh * DHEAD + ct * 16 + l16;
      At[((size_t)b * CN + pos) * CC + ch] = (__bf16)(acc[ct][r] * inv);
    }
  }
}

// ---------------------------------------------------------------------------
// Kernel 6: output projection + bias + residual (fp32 output).
// ---------------------------------------------------------------------------
__global__ __launch_bounds__(128) void proj_gemm(const __bf16* __restrict__ Wp,
                                                 const __bf16* __restrict__ At,
                                                 const float* __restrict__ bias,
                                                 const float* __restrict__ x,
                                                 float* __restrict__ out) {
  const int lane = threadIdx.x & 31, wave = threadIdx.x >> 5;
  const int half = lane >> 4, l16 = lane & 15;
  const int nbase = blockIdx.x * 16;                 // 144 position tiles
  const int obase = (blockIdx.y * 4 + wave) * 16;    // 32 channel tiles
  const int b = blockIdx.z;

  const __bf16* wrow = Wp + (size_t)(obase + l16) * CC;
  const __bf16* arow = At + ((size_t)b * CN + nbase + l16) * CC;
  v8f acc = {};
#pragma unroll
  for (int c = 0; c < CC; c += 32) {
    v16bf A = load_a_frag(wrow + c, half);
    v16bf B = *(const v16bf*)(arow + c + half * 16);
    acc = WMMA_BF16(A, B, acc);
  }
#pragma unroll
  for (int r = 0; r < 8; ++r) {
    int o = obase + r + 8 * half;
    int pos = nbase + l16;
    size_t idx = ((size_t)b * CC + o) * CN + pos;
    out[idx] = acc[r] + bias[o] + x[idx];
  }
}

// ---------------------------------------------------------------------------
extern "C" void kernel_launch(void* const* d_in, const int* in_sizes, int n_in,
                              void* d_out, int out_size, void* d_ws, size_t ws_size,
                              hipStream_t stream) {
  const float* x      = (const float*)d_in[0];
  const float* gn_w   = (const float*)d_in[1];
  const float* gn_b   = (const float*)d_in[2];
  const float* qkv_w  = (const float*)d_in[3];
  const float* qkv_b  = (const float*)d_in[4];
  const float* proj_w = (const float*)d_in[5];
  const float* proj_b = (const float*)d_in[6];
  float* out = (float*)d_out;

  char* ws = (char*)d_ws;
  size_t off = 0;
  auto take = [&](size_t bytes) -> char* {
    char* p = ws + off;
    off += (bytes + 255) & ~(size_t)255;
    return p;
  };
  float*  stats = (float*)take((size_t)CB * NGROUPS * 2 * sizeof(float));
  __bf16* Xt    = (__bf16*)take((size_t)CB * CN * CC * sizeof(__bf16));
  __bf16* Wq    = (__bf16*)take((size_t)3 * CC * CC * sizeof(__bf16));
  __bf16* Wp    = (__bf16*)take((size_t)CC * CC * sizeof(__bf16));
  __bf16* Qb    = (__bf16*)take((size_t)CB * NHEADS * CN * DHEAD * sizeof(__bf16));
  __bf16* Kb    = (__bf16*)take((size_t)CB * NHEADS * CN * DHEAD * sizeof(__bf16));
  __bf16* Vb    = (__bf16*)take((size_t)CB * NHEADS * DHEAD * CN * sizeof(__bf16));
  __bf16* At    = (__bf16*)take((size_t)CB * CN * CC * sizeof(__bf16));

  // 1+2) GroupNorm -> transposed bf16 activations
  gn_stats<<<CB * NGROUPS, 256, 0, stream>>>(x, stats);
  {
    size_t total = (size_t)CB * CN * CC;
    gn_apply<<<(int)((total + 255) / 256), 256, 0, stream>>>(x, gn_w, gn_b, stats, Xt);
  }
  // 3) weights to bf16
  cvt_w<<<(3 * CC * CC + 255) / 256, 256, 0, stream>>>(qkv_w, proj_w, Wq, Wp);
  // 4) QKV projection + head scatter
  qkv_gemm<<<dim3(CN / 16, (3 * CC) / 64, CB), 128, 0, stream>>>(Wq, Xt, qkv_b, Qb, Kb, Vb);
  // 5) flash attention per head (async LDS staging of K/V)
  attn<<<dim3((CN / 16) / 4, CB * NHEADS, 1), 128, 0, stream>>>(Qb, Kb, Vb, At);
  // 6) projection + bias + residual
  proj_gemm<<<dim3(CN / 16, CC / 64, CB), 128, 0, stream>>>(Wp, At, proj_b, x, out);
}
